// NeuralRecurrentDynamicsModel_34454227649266
// MI455X (gfx1250) — matrix operand order repaired and checked by reading, not compiled, not run on hardware
//
#include <hip/hip_runtime.h>
#include <hip/hip_bf16.h>
#include <math.h>

// ---------------------------------------------------------------------------
// Model dims
// ---------------------------------------------------------------------------
#define BSZ 256
#define HD  2048
#define ZD  4096
#define ASZ 32
#define PRE 6176          // ZD + HD + ASZ
#define PD  4096
#define SEQ 4
#define DM  1024
#define DI  2048
#define DS_ 16
#define DC_ 4
#define DTR 64
#define XPN (DTR + 2*DS_) // 96
#define ROWS (BSZ*SEQ)    // 1024

typedef __attribute__((ext_vector_type(16))) __bf16 v16bf;
typedef __attribute__((ext_vector_type(2)))  __bf16 bf16x2;
typedef __attribute__((ext_vector_type(8)))  float  v8f;

union Frag { unsigned int u[8]; v16bf v; };

// native HW converts (v_cvt_pk_bf16_f32 / v_cvt_bf16_f32), not manual RNE bit math
__device__ __forceinline__ unsigned short f2bf(float f) {
    __bf16 h = (__bf16)f;
    return __builtin_bit_cast(unsigned short, h);
}
__device__ __forceinline__ unsigned int pack2(float a, float b) {
    bf16x2 t;
    t.x = (__bf16)a;
    t.y = (__bf16)b;
    return __builtin_bit_cast(unsigned int, t);
}
__device__ __forceinline__ float sigf(float x)  { return 1.0f / (1.0f + __expf(-x)); }
__device__ __forceinline__ float siluf(float x) { return x * sigf(x); }

// ---------------------------------------------------------------------------
// Tiled GEMM:  C[M,N] = act( A[M,K] * B[K,N] + bias[N] )
// fp32 HBM -> bf16 LDS (HW packed converts) -> v_wmma_f32_16x16x32_bf16.
// Block tile 64x128x64, 8 waves (wave32), wave tile 32x32 (2x2 frags x 2 K-steps).
// Assumes M % 64 == 0 (true for all call sites). N / K tails handled via
// block-uniform slow path (only the xproj N=96 block and PRE=6176 K-tail).
// act: 0 = none, 1 = relu, 2 = softplus
// ---------------------------------------------------------------------------
__global__ __launch_bounds__(256)
void gemm_bf16_wmma(const float* __restrict__ A, int lda,
                    const float* __restrict__ Bw, int ldb,
                    const float* __restrict__ bias,
                    float* __restrict__ C, int ldc,
                    int M, int N, int K, int act)
{
    const int tid  = threadIdx.x;
    const int lane = tid & 31;
    const int wave = tid >> 5;
    const int wm   = wave >> 2;       // 0..1  (M direction)
    const int wn   = wave & 3;        // 0..3  (N direction)
    const int mBase = blockIdx.y * 64;
    const int nBase = blockIdx.x * 128;
    const bool nFull = (nBase + 128 <= N);

    __shared__ unsigned short As[64][68];    // [m][k] bf16, stride 136B (8B-aligned rows)
    __shared__ unsigned short Bs[128][68];   // [n][k] bf16 (transposed)

    const v8f vzero = {0.f,0.f,0.f,0.f,0.f,0.f,0.f,0.f};
    v8f acc[2][2];
    acc[0][0] = vzero; acc[0][1] = vzero; acc[1][0] = vzero; acc[1][1] = vzero;

    float4 ra[4];     // 64x64 A slab: 4 quads/thread
    float4 rb[8];     // 64x128 B slab: 8 quads/thread

    // ---- stage global -> registers (block-uniform fast/slow split) ----
    auto loadRegs = [&](int kb) {
        const bool kFull = (kb + 64 <= K);
        if (kFull) {
            #pragma unroll
            for (int i = 0; i < 4; ++i) {
                int q = tid + (i << 8);              // 0..1023
                int m = q >> 4, kq = (q & 15) << 2;  // 16 quads per 64-wide row
                ra[i] = *(const float4*)(A + (size_t)(mBase + m) * lda + kb + kq);
            }
        } else {
            #pragma unroll
            for (int i = 0; i < 4; ++i) {
                int q = tid + (i << 8);
                int m = q >> 4, kq = (q & 15) << 2;
                const float* ap = A + (size_t)(mBase + m) * lda;
                float v0 = (kb + kq + 0 < K) ? ap[kb + kq + 0] : 0.f;
                float v1 = (kb + kq + 1 < K) ? ap[kb + kq + 1] : 0.f;
                float v2 = (kb + kq + 2 < K) ? ap[kb + kq + 2] : 0.f;
                float v3 = (kb + kq + 3 < K) ? ap[kb + kq + 3] : 0.f;
                ra[i] = make_float4(v0, v1, v2, v3);
            }
        }
        if (kFull && nFull) {
            #pragma unroll
            for (int i = 0; i < 8; ++i) {
                int q = tid + (i << 8);              // 0..2047
                int k = q >> 5, nq = (q & 31) << 2;  // 32 quads per 128-wide row
                rb[i] = *(const float4*)(Bw + (size_t)(kb + k) * ldb + nBase + nq);
            }
        } else {
            #pragma unroll
            for (int i = 0; i < 8; ++i) {
                int q = tid + (i << 8);
                int k = q >> 5, nq = (q & 31) << 2;
                int gk = kb + k;
                const float* bp = Bw + (size_t)gk * ldb + nBase;
                bool kr = (gk < K);
                float v0 = (kr && nBase + nq + 0 < N) ? bp[nq + 0] : 0.f;
                float v1 = (kr && nBase + nq + 1 < N) ? bp[nq + 1] : 0.f;
                float v2 = (kr && nBase + nq + 2 < N) ? bp[nq + 2] : 0.f;
                float v3 = (kr && nBase + nq + 3 < N) ? bp[nq + 3] : 0.f;
                rb[i] = make_float4(v0, v1, v2, v3);
            }
        }
    };

    // ---- registers -> LDS (bf16 packed via v_cvt_pk_bf16_f32) ----
    auto storeRegs = [&]() {
        #pragma unroll
        for (int i = 0; i < 4; ++i) {
            int q = tid + (i << 8);
            int m = q >> 4, kq = (q & 15) << 2;
            uint2 p;
            p.x = pack2(ra[i].x, ra[i].y);
            p.y = pack2(ra[i].z, ra[i].w);
            *(uint2*)&As[m][kq] = p;                 // ds_store_b64, 8B-aligned
        }
        #pragma unroll
        for (int i = 0; i < 8; ++i) {
            int q = tid + (i << 8);
            int k = q >> 5, nq = (q & 31) << 2;
            Bs[nq + 0][k] = f2bf(rb[i].x);           // transpose: 4x ds_store_b16
            Bs[nq + 1][k] = f2bf(rb[i].y);
            Bs[nq + 2][k] = f2bf(rb[i].z);
            Bs[nq + 3][k] = f2bf(rb[i].w);
        }
    };

    loadRegs(0);
    const int g8  = (lane >> 4) << 3;
    const int g16 = (lane >> 4) << 4;

    for (int kb = 0; kb < K; kb += 64) {
        __syncthreads();                 // previous slab's consumers done
        storeRegs();
        __syncthreads();                 // LDS slab ready
        if (kb + 64 < K) loadRegs(kb + 64);   // overlap next global loads with math

        #pragma unroll
        for (int ks = 0; ks < 2; ++ks) {
            const int ko = ks << 5;
            Frag a[2], b[2];
            #pragma unroll
            for (int i = 0; i < 2; ++i) {
                int mrow = wm * 32 + i * 16 + (lane & 15);
                #pragma unroll
                for (int v = 0; v < 8; ++v) {
                    int kk = ko + g8 + ((v & 4) ? (8 + 2 * v) : (2 * v)); // ISA A-frag layout
                    a[i].u[v] = *(const unsigned int*)&As[mrow][kk];
                }
            }
            #pragma unroll
            for (int j = 0; j < 2; ++j) {
                int ncol = wn * 32 + j * 16 + (lane & 15);
                #pragma unroll
                for (int v = 0; v < 8; ++v)
                    b[j].u[v] = *(const unsigned int*)&Bs[ncol][ko + g8 + ((v & 4) ? (8 + 2 * v) : (2 * v))];
            }
            #pragma unroll
            for (int i = 0; i < 2; ++i)
                #pragma unroll
                for (int j = 0; j < 2; ++j)
                    acc[i][j] = __builtin_amdgcn_wmma_f32_16x16x32_bf16(
                        false, a[i].v, false, b[j].v, (short)0, acc[i][j], false, false);
        }
    }

    // ---- epilogue: bias + activation (C/D layout per ISA table) ----
    #pragma unroll
    for (int i = 0; i < 2; ++i) {
        #pragma unroll
        for (int j = 0; j < 2; ++j) {
            int mSub  = mBase + wm * 32 + i * 16 + ((lane >> 4) << 3);
            int nGlob = nBase + wn * 32 + j * 16 + (lane & 15);
            if (!nFull && nGlob >= N) continue;
            float bv = bias ? bias[nGlob] : 0.f;
            #pragma unroll
            for (int r = 0; r < 8; ++r) {
                float v = acc[i][j][r] + bv;
                if (act == 1)      v = v > 0.f ? v : 0.f;
                else if (act == 2) v = (v > 20.f) ? v : log1pf(__expf(v));
                C[(size_t)(mSub + r) * ldc + nGlob] = v;
            }
        }
    }
}

// ---------------------------------------------------------------------------
// Elementwise kernels
// ---------------------------------------------------------------------------
__global__ void concat_kernel(const float* __restrict__ obs, const float* __restrict__ h,
                              const float* __restrict__ a, float* __restrict__ X)
{
    int idx = blockIdx.x * blockDim.x + threadIdx.x;
    if (idx >= BSZ * PRE) return;
    int r = idx / PRE, c = idx % PRE;
    float v;
    if (c < ZD)            v = obs[(size_t)r * ZD + c];
    else if (c < ZD + HD)  v = h[(size_t)r * HD + (c - ZD)];
    else                   v = a[(size_t)r * ASZ + (c - ZD - HD)];
    X[idx] = v;
}

// depthwise causal conv1d (L=SEQ=4, kernel DC=4) + bias + SiLU
__global__ void conv_silu_kernel(const float* __restrict__ xz,
                                 const float* __restrict__ cw,   // (DI, DC)
                                 const float* __restrict__ cb,   // (DI)
                                 float* __restrict__ xcs)
{
    int idx = blockIdx.x * blockDim.x + threadIdx.x;
    if (idx >= BSZ * DI) return;
    int b = idx >> 11, c = idx & (DI - 1);
    float xv[SEQ];
    #pragma unroll
    for (int l = 0; l < SEQ; ++l)
        xv[l] = xz[(size_t)(b * SEQ + l) * (2 * DI) + c];
    float w[DC_];
    #pragma unroll
    for (int k = 0; k < DC_; ++k) w[k] = cw[(size_t)c * DC_ + k];
    float bias = cb[c];
    #pragma unroll
    for (int l = 0; l < SEQ; ++l) {
        float s = bias;
        #pragma unroll
        for (int k = 0; k < DC_; ++k) {
            int src = l + k - (DC_ - 1);
            if (src >= 0) s += w[k] * xv[src];
        }
        xcs[(size_t)(b * SEQ + l) * DI + c] = siluf(s);
    }
}

// selective scan: one thread per (batch, channel); L=4, DS=16 states in regs
__global__ void scan_kernel(const float* __restrict__ dt,    // (ROWS, DI)
                            const float* __restrict__ dbl,   // (ROWS, XPN)
                            const float* __restrict__ xcs,   // (ROWS, DI)
                            const float* __restrict__ xz,    // (ROWS, 2*DI) z gate
                            const float* __restrict__ Alog,  // (DI, DS)
                            const float* __restrict__ Dp,    // (DI)
                            float* __restrict__ yout)        // (ROWS, DI)
{
    int idx = blockIdx.x * blockDim.x + threadIdx.x;
    if (idx >= BSZ * DI) return;
    int b = idx >> 11, d = idx & (DI - 1);
    float negA[DS_], hst[DS_];
    #pragma unroll
    for (int s = 0; s < DS_; ++s) { negA[s] = -__expf(Alog[(size_t)d * DS_ + s]); hst[s] = 0.f; }
    float Dd = Dp[d];
    for (int l = 0; l < SEQ; ++l) {
        int r = b * SEQ + l;
        float dtv = dt[(size_t)r * DI + d];
        float xcv = xcs[(size_t)r * DI + d];
        float y = 0.f;
        #pragma unroll
        for (int s = 0; s < DS_; ++s) {
            float Bm = dbl[(size_t)r * XPN + DTR + s];
            float Cm = dbl[(size_t)r * XPN + DTR + DS_ + s];
            float dA = __expf(dtv * negA[s]);
            hst[s] = dA * hst[s] + dtv * Bm * xcv;
            y += hst[s] * Cm;
        }
        y += Dd * xcv;
        float zv = xz[(size_t)r * (2 * DI) + DI + d];
        yout[(size_t)r * DI + d] = y * siluf(zv);
    }
}

__global__ void gru_combine_kernel(const float* __restrict__ gi, const float* __restrict__ gh,
                                   const float* __restrict__ h, float* __restrict__ out)
{
    int idx = blockIdx.x * blockDim.x + threadIdx.x;
    if (idx >= BSZ * HD) return;
    int b = idx >> 11, j = idx & (HD - 1);
    const float* gib = gi + (size_t)b * 3 * HD;
    const float* ghb = gh + (size_t)b * 3 * HD;
    float r  = sigf(gib[j]          + ghb[j]);
    float zt = sigf(gib[HD + j]     + ghb[HD + j]);
    float n  = tanhf(gib[2 * HD + j] + r * ghb[2 * HD + j]);
    out[idx] = (1.f - zt) * n + zt * h[idx];
}

__device__ __forceinline__ float u01(unsigned int x) {
    x ^= x >> 17; x *= 0xed5ad4bbu; x ^= x >> 11; x *= 0xac4c1b51u;
    x ^= x >> 15; x *= 0x31848babu; x ^= x >> 14;
    return (float)(x >> 8) * (1.0f / 16777216.0f) + 1e-7f;
}

// rowwise softmax over groups of 32 + Gumbel-argmax one-hot (ST forward == one-hot)
__global__ void softmax_sample_kernel(const float* __restrict__ logits,
                                      float* __restrict__ sample_out,
                                      float* __restrict__ prob_out)
{
    int row = blockIdx.x * blockDim.x + threadIdx.x;
    if (row >= BSZ * ZD / 32) return;
    const float* Lr = logits + (size_t)row * 32;
    float mx = -INFINITY;
    #pragma unroll
    for (int i = 0; i < 32; ++i) mx = fmaxf(mx, Lr[i]);
    float e[32], s = 0.f;
    #pragma unroll
    for (int i = 0; i < 32; ++i) { e[i] = __expf(Lr[i] - mx); s += e[i]; }
    float inv = 1.f / s;
    int arg = 0; float best = -INFINITY;
    #pragma unroll
    for (int i = 0; i < 32; ++i) {
        prob_out[(size_t)row * 32 + i] = e[i] * inv;
        float u = u01((unsigned int)(row * 32 + i) * 2654435761u + 42u);
        float g = (Lr[i] - mx) - logf(-logf(u));
        if (g > best) { best = g; arg = i; }
    }
    #pragma unroll
    for (int i = 0; i < 32; ++i)
        sample_out[(size_t)row * 32 + i] = (i == arg) ? 1.f : 0.f;
}

// ---------------------------------------------------------------------------
// Host orchestration
// ---------------------------------------------------------------------------
static inline void gemm(hipStream_t s, const float* A, int lda, const float* Bw, int ldb,
                        const float* bias, float* C, int ldc, int M, int N, int K, int act)
{
    dim3 grid((N + 127) / 128, (M + 63) / 64);
    gemm_bf16_wmma<<<grid, 256, 0, s>>>(A, lda, Bw, ldb, bias, C, ldc, M, N, K, act);
}

extern "C" void kernel_launch(void* const* d_in, const int* in_sizes, int n_in,
                              void* d_out, int out_size, void* d_ws, size_t ws_size,
                              hipStream_t stream)
{
    (void)in_sizes; (void)n_in; (void)out_size; (void)ws_size;
    const float* obs = (const float*)d_in[0];
    const float* hst = (const float*)d_in[1];
    const float* actn = (const float*)d_in[2];
    const float* m0_w0 = (const float*)d_in[3];  const float* m0_b0 = (const float*)d_in[4];
    const float* m0_w1 = (const float*)d_in[5];  const float* m0_b1 = (const float*)d_in[6];
    const float* gru_wih = (const float*)d_in[7]; const float* gru_bih = (const float*)d_in[8];
    const float* gru_whh = (const float*)d_in[9]; const float* gru_bhh = (const float*)d_in[10];
    const float* mz_w0 = (const float*)d_in[11]; const float* mz_b0 = (const float*)d_in[12];
    const float* mz_w1 = (const float*)d_in[13]; const float* mz_b1 = (const float*)d_in[14];
    const float* m1_w0 = (const float*)d_in[15]; const float* m1_b0 = (const float*)d_in[16];
    const float* m1_w1 = (const float*)d_in[17]; const float* m1_b1 = (const float*)d_in[18];
    // mamba param blocks: [in_w, conv_w, conv_b, xproj_w, dt_w, dt_b, A_log, D, out_w]
    const float* const* mb1 = (const float* const*)&d_in[19];
    const float* const* mb2 = (const float* const*)&d_in[28];

    float* ws = (float*)d_ws;
    float* X    = ws;                          // 256*6176
    float* bufA = X    + (size_t)BSZ * PRE;    // 256*4096
    float* bufB = bufA + (size_t)BSZ * PD;     // 256*4096
    float* bufC = bufB + (size_t)BSZ * PD;     // 256*4096 (gru_x / x2)
    float* XZ   = bufC + (size_t)BSZ * PD;     // 1024*4096
    float* XCS  = XZ   + (size_t)ROWS * 2*DI;  // 1024*2048
    float* DBL  = XCS  + (size_t)ROWS * DI;    // 1024*96
    float* DT   = DBL  + (size_t)ROWS * XPN;   // 1024*2048
    float* Y    = DT   + (size_t)ROWS * DI;    // 1024*2048
    float* GI   = Y    + (size_t)ROWS * DI;    // 256*6144
    float* GH   = GI   + (size_t)BSZ * 3*HD;   // 256*6144

    float* out_sample = (float*)d_out;                 // (B, Z)
    float* out_prob   = out_sample + (size_t)BSZ * ZD; // (B, Z)
    float* out_h      = out_prob   + (size_t)BSZ * ZD; // (B, H)

    const int EW = 256;
    // 1) concat -> X (256, 6176)
    concat_kernel<<<(BSZ * PRE + EW - 1) / EW, EW, 0, stream>>>(obs, hst, actn, X);
    // 2) mlp_0
    gemm(stream, X,    PRE, m0_w0, PD, m0_b0, bufA, PD, BSZ, PD, PRE, 1);
    gemm(stream, bufA, PD,  m0_w1, PD, m0_b1, bufB, PD, BSZ, PD, PD,  0);
    // 3) mamba block 1 on bufB viewed as (1024, 1024)
    gemm(stream, bufB, DM, mb1[0], 2 * DI, nullptr, XZ, 2 * DI, ROWS, 2 * DI, DM, 0);
    conv_silu_kernel<<<(BSZ * DI + EW - 1) / EW, EW, 0, stream>>>(XZ, mb1[1], mb1[2], XCS);
    gemm(stream, XCS, DI, mb1[3], XPN, nullptr, DBL, XPN, ROWS, XPN, DI, 0);
    gemm(stream, DBL, XPN, mb1[4], DI, mb1[5], DT, DI, ROWS, DI, DTR, 2);   // softplus
    scan_kernel<<<(BSZ * DI + EW - 1) / EW, EW, 0, stream>>>(DT, DBL, XCS, XZ, mb1[6], mb1[7], Y);
    gemm(stream, Y, DI, mb1[8], DM, nullptr, bufC, DM, ROWS, DM, DI, 0);    // gru_x (256,4096)
    // 4) GRU cell
    gemm(stream, bufC, PD, gru_wih, 3 * HD, gru_bih, GI, 3 * HD, BSZ, 3 * HD, PD, 0);
    gemm(stream, hst,  HD, gru_whh, 3 * HD, gru_bhh, GH, 3 * HD, BSZ, 3 * HD, HD, 0);
    gru_combine_kernel<<<(BSZ * HD + EW - 1) / EW, EW, 0, stream>>>(GI, GH, hst, out_h);
    // 5) pre_z_pred mlp
    gemm(stream, hst,  HD, mz_w0, PD, mz_b0, bufA, PD, BSZ, PD, HD, 1);
    gemm(stream, bufA, PD, mz_w1, PD, mz_b1, bufB, PD, BSZ, PD, PD, 0);
    // 6) mamba block 2
    gemm(stream, bufB, DM, mb2[0], 2 * DI, nullptr, XZ, 2 * DI, ROWS, 2 * DI, DM, 0);
    conv_silu_kernel<<<(BSZ * DI + EW - 1) / EW, EW, 0, stream>>>(XZ, mb2[1], mb2[2], XCS);
    gemm(stream, XCS, DI, mb2[3], XPN, nullptr, DBL, XPN, ROWS, XPN, DI, 0);
    gemm(stream, DBL, XPN, mb2[4], DI, mb2[5], DT, DI, ROWS, DI, DTR, 2);
    scan_kernel<<<(BSZ * DI + EW - 1) / EW, EW, 0, stream>>>(DT, DBL, XCS, XZ, mb2[6], mb2[7], Y);
    gemm(stream, Y, DI, mb2[8], DM, nullptr, bufC, DM, ROWS, DM, DI, 0);    // x2 (256,4096)
    // 7) mlp_1 -> logits
    gemm(stream, bufC, PD, m1_w0, ZD, m1_b0, bufA, ZD, BSZ, ZD, PD, 1);
    gemm(stream, bufA, ZD, m1_w1, ZD, m1_b1, bufB, ZD, BSZ, ZD, ZD, 0);
    // 8) softmax + straight-through sample
    softmax_sample_kernel<<<(BSZ * ZD / 32 + EW - 1) / EW, EW, 0, stream>>>(bufB, out_sample, out_prob);
}
